// Sublayers_27857157882269
// MI455X (gfx1250) — matrix operand order repaired and checked by reading, not compiled, not run on hardware
//
#include <hip/hip_runtime.h>

typedef _Float16 half_t;
typedef __attribute__((ext_vector_type(16))) _Float16 v16h;
typedef __attribute__((ext_vector_type(8)))  _Float16 v8h;
typedef __attribute__((ext_vector_type(8)))  float    v8f;

// ---------------------------------------------------------------------------
// WMMA helper (CDNA5 wave32, V_WMMA_F32_16X16X32_F16)
// ---------------------------------------------------------------------------
__device__ __forceinline__ v8f wmma_f16(v16h a, v16h b, v8f c) {
  return __builtin_amdgcn_wmma_f32_16x16x32_f16(
      /*neg_a=*/false, a, /*neg_b=*/false, b,
      /*c_mod=*/(short)0, c, /*reuse_a=*/false, /*reuse_b=*/false);
}

// ---------------------------------------------------------------------------
// Async Global->LDS copy (CDNA5, ASYNCcnt-tracked) + wait
// global_load_async_to_lds_b128 vdst(LDS byte offset), vaddr(64b global), off
// ---------------------------------------------------------------------------
__device__ __forceinline__ void async_copy_b128(const half_t* lds_dst,
                                                const half_t* gsrc) {
  uint32_t loff = (uint32_t)(uintptr_t)lds_dst;  // flat LDS addr low 32 = offset
  asm volatile("global_load_async_to_lds_b128 %0, %1, off"
               :: "v"(loff), "v"(gsrc)
               : "memory");
}

template <int N>
__device__ __forceinline__ void wait_asynccnt() {
  asm volatile("s_wait_asynccnt %0" :: "n"(N) : "memory");
}

// ---------------------------------------------------------------------------
// Fragment loaders, ISA 7.12.2 layouts.
// A 16x32 f16 (MxK row-major): lane<16: K0-7 (v0-3), K16-23 (v4-7);
//                              lane>=16: K8-15, K24-31; row = lane&15.
// B 32x16 f16 as N x K row-major: lane<16: K0-7, K16-23; lane>=16: K8-15, K24-31.
// ---------------------------------------------------------------------------
__device__ __forceinline__ v16h frag_a(const half_t* base, int ld, int lane) {
  const half_t* p = base + (lane & 15) * ld + ((lane >> 4) << 3);
  union { v16h v; v8h h[2]; } u;
  u.h[0] = *(const v8h*)(p);
  u.h[1] = *(const v8h*)(p + 16);
  return u.v;
}
__device__ __forceinline__ v16h frag_b(const half_t* base, int ld, int lane) {
  const half_t* p = base + (lane & 15) * ld + ((lane >> 4) << 3);
  union { v16h v; v8h h[2]; } u;
  u.h[0] = *(const v8h*)(p);
  u.h[1] = *(const v8h*)(p + 16);
  return u.v;
}

// ---------------------------------------------------------------------------
// Elementwise helpers
// ---------------------------------------------------------------------------
__global__ __launch_bounds__(256) void cvt_f32_f16(const float* __restrict__ in,
                                                   half_t* __restrict__ out, int n) {
  int i = blockIdx.x * 256 + threadIdx.x;
  if (i < n) out[i] = (half_t)in[i];
}

__global__ __launch_bounds__(256) void zero_f32(float* __restrict__ p, int n) {
  int i = blockIdx.x * 256 + threadIdx.x;
  if (i < n) p[i] = 0.0f;
}

// ---------------------------------------------------------------------------
// LayerNorm: one block per row, D=1024, ddof=1 variance, /(std+eps), f16 out.
// ---------------------------------------------------------------------------
__global__ __launch_bounds__(256) void ln_kernel(const float* __restrict__ in,
                                                 const float* __restrict__ ga,
                                                 const float* __restrict__ gb,
                                                 half_t* __restrict__ out) {
  const int D = 1024;
  __shared__ float red[256];
  int row = blockIdx.x;
  int t = threadIdx.x;
  const float* xr = in + (size_t)row * D;

  float s = 0.0f;
  for (int i = t; i < D; i += 256) s += xr[i];
  red[t] = s;
  __syncthreads();
  for (int o = 128; o > 0; o >>= 1) {
    if (t < o) red[t] += red[t + o];
    __syncthreads();
  }
  float mean = red[0] * (1.0f / D);
  __syncthreads();

  float ss = 0.0f;
  for (int i = t; i < D; i += 256) { float d = xr[i] - mean; ss += d * d; }
  red[t] = ss;
  __syncthreads();
  for (int o = 128; o > 0; o >>= 1) {
    if (t < o) red[t] += red[t + o];
    __syncthreads();
  }
  float stdv = sqrtf(red[0] * (1.0f / (D - 1)));
  float inv = 1.0f / (stdv + 1e-6f);

  half_t* orow = out + (size_t)row * D;
  for (int i = t; i < D; i += 256)
    orow[i] = (half_t)(ga[i] * (xr[i] - mean) * inv + gb[i]);
}

// ---------------------------------------------------------------------------
// WMMA GEMM with async double-buffered LDS staging.
//   C[M,N] = op( A_f16[M,K] @ W_f16[N,K]^T + bias[N] )
// Block 256 thr = 8 waves; block tile 128x128; wave tile 32x64 (2x4 WMMA).
// Stage: A slab 128x32, B slab 128x32 (f16, rows padded to 40 halfs / 80B to
// break LDS bank conflicts), double buffered = 40KB LDS.
// Copies: 4x global_load_async_to_lds_b128 per thread per stage.
// MODE 0: f16 out [M,N]
// MODE 1: f16 out [M,N], ReLU
// MODE 2: f32 out [M,N], += residual[M,N]
// MODE 3: f16 out transposed [N,M]   (used to store V^T)
// ---------------------------------------------------------------------------
#define ROWH 40   // padded LDS row length in halfs (32 data + 8 pad)

template <int MODE>
__global__ __launch_bounds__(256) void gemm_kernel(const half_t* __restrict__ A,
                                                   const half_t* __restrict__ W,
                                                   const float* __restrict__ bias,
                                                   const float* __restrict__ resid,
                                                   void* __restrict__ outp,
                                                   int M, int N, int K) {
  __shared__ half_t sA[2][128 * ROWH];
  __shared__ half_t sB[2][128 * ROWH];

  int tid  = threadIdx.x;
  int lane = tid & 31;
  int wave = tid >> 5;
  int row0 = blockIdx.y * 128;
  int col0 = blockIdx.x * 128;
  int wm = (wave & 3) * 32;   // 4 waves tile M
  int wn = (wave >> 2) * 64;  // 2 waves tile N

  // issue one stage (A slab + B slab) of async copies into buffer `buf`
  auto issue_stage = [&](int kk, int buf) {
#pragma unroll
    for (int i = 0; i < 2; ++i) {
      int c = tid + 256 * i;          // chunk 0..511, 16B each
      int r = c >> 2, seg = c & 3;    // row, 16B segment within 64B row
      async_copy_b128(&sA[buf][r * ROWH + seg * 8],
                      A + (size_t)(row0 + r) * K + kk + seg * 8);
    }
#pragma unroll
    for (int i = 0; i < 2; ++i) {
      int c = tid + 256 * i;
      int r = c >> 2, seg = c & 3;
      async_copy_b128(&sB[buf][r * ROWH + seg * 8],
                      W + (size_t)(col0 + r) * K + kk + seg * 8);
    }
  };

  int nk = K / 32;
  issue_stage(0, 0);
  if (nk > 1) issue_stage(32, 1);

  v8f acc[2][4] = {};
  for (int k = 0; k < nk; ++k) {
    if (k + 1 < nk) wait_asynccnt<4>();   // stage k retired, k+1 in flight
    else            wait_asynccnt<0>();
    __syncthreads();

    int buf = k & 1;
    v16h a0 = frag_a(&sA[buf][(wm +  0) * ROWH], ROWH, lane);
    v16h a1 = frag_a(&sA[buf][(wm + 16) * ROWH], ROWH, lane);
    v16h b0 = frag_b(&sB[buf][(wn +  0) * ROWH], ROWH, lane);
    v16h b1 = frag_b(&sB[buf][(wn + 16) * ROWH], ROWH, lane);
    v16h b2 = frag_b(&sB[buf][(wn + 32) * ROWH], ROWH, lane);
    v16h b3 = frag_b(&sB[buf][(wn + 48) * ROWH], ROWH, lane);

    acc[0][0] = wmma_f16(a0, b0, acc[0][0]);
    acc[0][1] = wmma_f16(a0, b1, acc[0][1]);
    acc[0][2] = wmma_f16(a0, b2, acc[0][2]);
    acc[0][3] = wmma_f16(a0, b3, acc[0][3]);
    acc[1][0] = wmma_f16(a1, b0, acc[1][0]);
    acc[1][1] = wmma_f16(a1, b1, acc[1][1]);
    acc[1][2] = wmma_f16(a1, b2, acc[1][2]);
    acc[1][3] = wmma_f16(a1, b3, acc[1][3]);

    __syncthreads();                      // stage k fully consumed block-wide
    if (k + 2 < nk) issue_stage((k + 2) * 32, buf);
  }

  int hi8 = (lane >> 4) << 3;   // C/D layout: lane>=16 -> rows +8
  int c15 = lane & 15;
#pragma unroll
  for (int mi = 0; mi < 2; ++mi)
#pragma unroll
    for (int ni = 0; ni < 4; ++ni)
#pragma unroll
      for (int j = 0; j < 8; ++j) {
        int r = row0 + wm + mi * 16 + j + hi8;
        int c = col0 + wn + ni * 16 + c15;
        float v = acc[mi][ni][j] + bias[c];
        if (MODE == 1) v = fmaxf(v, 0.0f);
        if (MODE == 2) {
          ((float*)outp)[(size_t)r * N + c] = v + resid[(size_t)r * N + c];
        } else if (MODE == 3) {
          ((half_t*)outp)[(size_t)c * M + r] = (half_t)v;
        } else {
          ((half_t*)outp)[(size_t)r * N + c] = (half_t)v;
        }
      }
}

// ---------------------------------------------------------------------------
// Attention: grid (S/32, B); block 256 (8 waves); loops over all 16 heads so
// weight_norm rows are exclusively owned by this block (deterministic, no
// atomics). Dynamic LDS: scores f32[32][1024] (128KB) + probs f16[32][1024].
// ---------------------------------------------------------------------------
__global__ __launch_bounds__(256) void attn_kernel(const half_t* __restrict__ Q,
                                                   const half_t* __restrict__ Kb,
                                                   const half_t* __restrict__ Vt,
                                                   const int* __restrict__ mask,
                                                   half_t* __restrict__ attnO,
                                                   float* __restrict__ wnorm) {
  extern __shared__ char smem[];
  const int S = 1024, D = 1024, H = 16, DH = 64;
  float*  sc = (float*)smem;                              // [32][S]
  half_t* pr = (half_t*)(smem + (size_t)32 * S * 4);      // [32][S]

  int lane = threadIdx.x & 31;
  int wave = threadIdx.x >> 5;
  int hi8 = (lane >> 4) << 3;
  int c15 = lane & 15;
  int q0 = blockIdx.x * 32;
  int b  = blockIdx.y;

  for (int h = 0; h < H; ++h) {
    // ---- phase 1: scores = Q K^T / sqrt(DH), mask -> LDS -------------------
    for (int t = wave; t < 128; t += 8) {          // 2 q-subtiles x 64 k-tiles
      int qs = (t & 1) * 16;
      int k0 = (t >> 1) * 16;
      v8f acc = {};
#pragma unroll
      for (int kk = 0; kk < DH; kk += 32) {
        v16h a  = frag_a(Q  + (size_t)(b * S + q0 + qs) * D + h * DH + kk, D, lane);
        v16h bm = frag_b(Kb + (size_t)(b * S + k0) * D + h * DH + kk, D, lane);
        acc = wmma_f16(a, bm, acc);
      }
      int col = k0 + c15;
      int mk = mask[b * S + col];
#pragma unroll
      for (int j = 0; j < 8; ++j) {
        float v = acc[j] * 0.125f;                 // 1/sqrt(64)
        sc[(qs + j + hi8) * S + col] = mk ? v : -1e9f;
      }
    }
    __syncthreads();

    // ---- phase 2: row softmax + weight_norm accumulation -------------------
    for (int r = wave; r < 32; r += 8) {
      float mx = -3.4e38f;
      for (int j = lane; j < S; j += 32) mx = fmaxf(mx, sc[r * S + j]);
      for (int o = 16; o > 0; o >>= 1) mx = fmaxf(mx, __shfl_xor(mx, o, 32));
      float sum = 0.0f;
      for (int j = lane; j < S; j += 32) sum += __expf(sc[r * S + j] - mx);
      for (int o = 16; o > 0; o >>= 1) sum += __shfl_xor(sum, o, 32);
      float inv = 1.0f / sum;
      float* wn = wnorm + (size_t)(b * S + q0 + r) * S;
      for (int j = lane; j < S; j += 32) {
        float p = __expf(sc[r * S + j] - mx) * inv;
        pr[r * S + j] = (half_t)p;
        wn[j] += p * 0.0625f;                      // 1/N_HEADS; block-exclusive
      }
    }
    __syncthreads();

    // ---- phase 3: out = P @ V (V stored transposed [D, B*S]) ---------------
    {
      int qs = (wave >> 2) * 16;                   // 2 q-subtiles
      int nt = (wave & 3) * 16;                    // 4 dh-tiles (DH=64)
      v8f acc = {};
      for (int kk = 0; kk < S; kk += 32) {
        v16h a  = frag_a(pr + (size_t)qs * S + kk, S, lane);
        v16h bm = frag_b(Vt + (size_t)(h * DH + nt) * (4 * S) + b * S + kk,
                         4 * S, lane);
        acc = wmma_f16(a, bm, acc);
      }
#pragma unroll
      for (int j = 0; j < 8; ++j) {
        int r = q0 + qs + j + hi8;
        int c = h * DH + nt + c15;
        attnO[(size_t)(b * S + r) * D + c] = (half_t)acc[j];
      }
    }
    __syncthreads();                               // sc reused next head
  }
}

// ---------------------------------------------------------------------------
// Host-side orchestration
// ---------------------------------------------------------------------------
extern "C" void kernel_launch(void* const* d_in, const int* in_sizes, int n_in,
                              void* d_out, int out_size, void* d_ws, size_t ws_size,
                              hipStream_t stream) {
  (void)in_sizes; (void)n_in; (void)out_size; (void)ws_size;

  const float* x    = (const float*)d_in[0];
  const int*   mask = (const int*)  d_in[1];
  const float* ln_a = (const float*)d_in[2];
  const float* ln_b = (const float*)d_in[3];
  const float* Wq   = (const float*)d_in[4];
  const float* bq   = (const float*)d_in[5];
  const float* Wk   = (const float*)d_in[6];
  const float* bk   = (const float*)d_in[7];
  const float* Wv   = (const float*)d_in[8];
  const float* bv   = (const float*)d_in[9];
  const float* Wo   = (const float*)d_in[10];
  const float* bo   = (const float*)d_in[11];
  const float* W1   = (const float*)d_in[12];
  const float* b1   = (const float*)d_in[13];
  const float* W2   = (const float*)d_in[14];
  const float* b2   = (const float*)d_in[15];

  const int B = 4, S = 1024, D = 1024, FF = 4096;
  const int R = B * S;  // 4096 rows

  // workspace carve-up (256B aligned)
  char* w = (char*)d_ws;
  auto carve = [&](size_t bytes) -> char* {
    char* p = w;
    w += (bytes + 255) & ~(size_t)255;
    return p;
  };
  half_t* xn    = (half_t*)carve((size_t)R * D * 2);
  half_t* Wq16  = (half_t*)carve((size_t)D * D * 2);
  half_t* Wk16  = (half_t*)carve((size_t)D * D * 2);
  half_t* Wv16  = (half_t*)carve((size_t)D * D * 2);
  half_t* Wo16  = (half_t*)carve((size_t)D * D * 2);
  half_t* W116  = (half_t*)carve((size_t)FF * D * 2);
  half_t* W216  = (half_t*)carve((size_t)D * FF * 2);
  half_t* Qf    = (half_t*)carve((size_t)R * D * 2);
  half_t* Kf    = (half_t*)carve((size_t)R * D * 2);
  half_t* Vt    = (half_t*)carve((size_t)D * R * 2);   // transposed [D, R]
  half_t* attnO = (half_t*)carve((size_t)R * D * 2);
  float*  Z1    = (float*) carve((size_t)R * D * 4);
  half_t* zn    = (half_t*)carve((size_t)R * D * 2);
  half_t* hbuf  = (half_t*)carve((size_t)R * FF * 2);

  float* Z2    = (float*)d_out;                        // [B,S,D]
  float* wnorm = (float*)d_out + (size_t)B * S * D;    // [B,S,S]

  // 1) weights -> f16
  int nDD = D * D, nFD = FF * D;
  cvt_f32_f16<<<dim3((nDD + 255) / 256), 256, 0, stream>>>(Wq, Wq16, nDD);
  cvt_f32_f16<<<dim3((nDD + 255) / 256), 256, 0, stream>>>(Wk, Wk16, nDD);
  cvt_f32_f16<<<dim3((nDD + 255) / 256), 256, 0, stream>>>(Wv, Wv16, nDD);
  cvt_f32_f16<<<dim3((nDD + 255) / 256), 256, 0, stream>>>(Wo, Wo16, nDD);
  cvt_f32_f16<<<dim3((nFD + 255) / 256), 256, 0, stream>>>(W1, W116, nFD);
  cvt_f32_f16<<<dim3((nFD + 255) / 256), 256, 0, stream>>>(W2, W216, nFD);

  // 2) zero weight_norm output region
  zero_f32<<<dim3(((size_t)B * S * S + 255) / 256), 256, 0, stream>>>(wnorm, B * S * S);

  // 3) pre-norm
  ln_kernel<<<dim3(R), 256, 0, stream>>>(x, ln_a, ln_b, xn);

  // 4) Q, K, V projections (V stored transposed)
  dim3 gDD(D / 128, R / 128);
  gemm_kernel<0><<<gDD, 256, 0, stream>>>(xn, Wq16, bq, nullptr, Qf, R, D, D);
  gemm_kernel<0><<<gDD, 256, 0, stream>>>(xn, Wk16, bk, nullptr, Kf, R, D, D);
  gemm_kernel<3><<<gDD, 256, 0, stream>>>(xn, Wv16, bv, nullptr, Vt, R, D, D);

  // 5) attention (scores+softmax+PV+weight_norm), 192KB dynamic LDS
  size_t attn_lds = (size_t)32 * S * 4 + (size_t)32 * S * 2;
  attn_kernel<<<dim3(S / 32, B), 256, attn_lds, stream>>>(Qf, Kf, Vt, mask, attnO, wnorm);

  // 6) output projection + residual: Z1 = x + attnO @ Wo^T + bo
  gemm_kernel<2><<<gDD, 256, 0, stream>>>(attnO, Wo16, bo, x, Z1, R, D, D);

  // 7) pre-norm for FFN
  ln_kernel<<<dim3(R), 256, 0, stream>>>(Z1, ln_a, ln_b, zn);

  // 8) FFN: h = relu(zn @ W1^T + b1)
  gemm_kernel<1><<<dim3(FF / 128, R / 128), 256, 0, stream>>>(zn, W116, b1, nullptr, hbuf, R, FF, D);

  // 9) Z2 = Z1 + h @ W2^T + b2  -> d_out
  gemm_kernel<2><<<gDD, 256, 0, stream>>>(hbuf, W216, b2, Z1, Z2, R, D, FF);
}
